// MultiHeadCausalAttention_67070209294919
// MI455X (gfx1250) — matrix operand order repaired and checked by reading, not compiled, not run on hardware
//
#include <hip/hip_runtime.h>
#include <hip/hip_bf16.h>

#define EMBED    2048
#define HEADS    16
#define HEAD_DIM 128
#define SEQ      2048
#define BATCH    2
#define MTOT     (BATCH * SEQ)   // 4096 rows for all GEMMs

typedef __attribute__((ext_vector_type(16))) __bf16 v16bf;
typedef __attribute__((ext_vector_type(8)))  __bf16 v8bf;
typedef __attribute__((ext_vector_type(8)))  float  v8f;

union BF16x16 { v16bf v; v8bf h[2]; };

__device__ inline v8f zero8() {
  v8f z;
#pragma unroll
  for (int i = 0; i < 8; ++i) z[i] = 0.f;
  return z;
}

__device__ inline v8f wmma_bf16(v16bf a, v16bf b, v8f c) {
  // D = A(16x32 bf16) * B(32x16 bf16) + C(16x16 f32)
  return __builtin_amdgcn_wmma_f32_16x16x32_bf16(
      /*neg_a=*/false, a, /*neg_b=*/false, b,
      /*c_mod=*/(short)0, c, /*reuse_a=*/false, /*reuse_b=*/false);
}

// ---------------------------------------------------------------------------
// fp32 -> bf16 conversion, 8 elements per thread-iteration (b128 store)
// ---------------------------------------------------------------------------
__global__ void f32_to_bf16_kernel(const float* __restrict__ in,
                                   __bf16* __restrict__ out, int n8) {
  int i = blockIdx.x * blockDim.x + threadIdx.x;
  const int stride = gridDim.x * blockDim.x;
  for (; i < n8; i += stride) {
    const float* p = in + (size_t)i * 8;
    v8bf o;
#pragma unroll
    for (int j = 0; j < 8; ++j) o[j] = (__bf16)p[j];
    *(v8bf*)(out + (size_t)i * 8) = o;
  }
}

// ---------------------------------------------------------------------------
// NT GEMM:  C[M,N] = A[M,K](bf16,row-major) * W[N,K](bf16,row-major)^T + bias
// Each wave computes a 64x64 tile (4 M-frags x 4 N-frags): 16 WMMA per
// 32-deep K step against 16 b128 loads (1.5x the intensity of a 32x64 tile).
// mode 0: write bf16 to [B,H,S,D] (QKV path). mode 1: write f32 row-major.
// ---------------------------------------------------------------------------
__global__ __launch_bounds__(128)
void gemm_bf16_nt(const __bf16* __restrict__ A, const __bf16* __restrict__ W,
                  const float* __restrict__ bias, void* __restrict__ outp,
                  int M, int N, int K, int mode) {
  const int lane = threadIdx.x & 31;
  const int hlf  = lane >> 4;      // 0/1 half of wave
  const int rln  = lane & 15;
  const int tilesN = N >> 6;
  const int waveId = blockIdx.x * (blockDim.x >> 5) + (threadIdx.x >> 5);
  const int tm = waveId / tilesN;
  const int tn = waveId % tilesN;
  if (tm >= (M >> 6)) return;      // wave-uniform
  const int m0 = tm << 6;
  const int n0 = tn << 6;

  v8f acc[4][4];
#pragma unroll
  for (int mf = 0; mf < 4; ++mf)
#pragma unroll
    for (int nf = 0; nf < 4; ++nf) acc[mf][nf] = zero8();

  const __bf16* Arow = A + (size_t)(m0 + rln) * K;
  const __bf16* Wrow = W + (size_t)(n0 + rln) * K;

  for (int k0 = 0; k0 < K; k0 += 32) {
    __builtin_prefetch(Arow + k0 + 256, 0, 1);            // global_prefetch_b8
    __builtin_prefetch(Wrow + k0 + 256, 0, 1);

    BF16x16 a[4];
#pragma unroll
    for (int mf = 0; mf < 4; ++mf) {
      const __bf16* p = Arow + (size_t)mf * 16 * K + k0 + hlf * 16;
      a[mf].h[0] = *(const v8bf*)(p);        // K = k0 + hlf*16 .. +7
      a[mf].h[1] = *(const v8bf*)(p + 8);    // K = k0 + hlf*16 + 8 .. +15
    }
#pragma unroll
    for (int nf = 0; nf < 4; ++nf) {
      const __bf16* p = Wrow + (size_t)nf * 16 * K + k0 + hlf * 16;
      BF16x16 b;
      b.h[0] = *(const v8bf*)(p);            // K = k0 + hlf*16 .. +7
      b.h[1] = *(const v8bf*)(p + 8);        // K = k0 + hlf*16 + 8 .. +15
#pragma unroll
      for (int mf = 0; mf < 4; ++mf)
        acc[mf][nf] = wmma_bf16(a[mf].v, b.v, acc[mf][nf]);
    }
  }

#pragma unroll
  for (int mf = 0; mf < 4; ++mf)
#pragma unroll
    for (int nf = 0; nf < 4; ++nf)
#pragma unroll
      for (int r = 0; r < 8; ++r) {
        const int m = m0 + mf * 16 + hlf * 8 + r;
        const int n = n0 + nf * 16 + rln;
        const float v = acc[mf][nf][r] + bias[n];
        if (mode == 0) {
          // scatter to [B, H, S, D]
          const int b = m >> 11, s = m & (SEQ - 1);
          const int h = n >> 7, d = n & (HEAD_DIM - 1);
          ((__bf16*)outp)[(((size_t)(b * HEADS + h)) * SEQ + s) * HEAD_DIM + d] =
              (__bf16)v;
        } else {
          ((float*)outp)[(size_t)m * N + n] = v;
        }
      }
}

// ---------------------------------------------------------------------------
// Flash attention (causal), bf16 in/out, fp32 online softmax.
// Block = 128 thr (4 waves); each wave owns 16 query rows; block covers 64.
// grid = (SEQ/64, BATCH*HEADS)
// ---------------------------------------------------------------------------
__global__ __launch_bounds__(128)
void flash_attn(const __bf16* __restrict__ Q, const __bf16* __restrict__ Km,
                const __bf16* __restrict__ Vm, __bf16* __restrict__ outA) {
  __shared__ __align__(16) __bf16 Kt[32 * HEAD_DIM];   // [key][d]
  __shared__ __align__(16) __bf16 Vt[HEAD_DIM * 32];   // [d][key] (transposed)
  __shared__ __align__(16) __bf16 Pb[4][16 * 32];      // per-wave P scratch

  const int tid  = threadIdx.x;
  const int wave = tid >> 5;
  const int lane = tid & 31;
  const int hlf  = lane >> 4;
  const int rln  = lane & 15;
  const int bh   = blockIdx.y;                  // b*HEADS + h
  const int qblk = blockIdx.x;
  const int q0   = qblk * 64 + wave * 16;       // first query row of this wave

  // preload Q A-fragments (16 rows x 128 d, 4 K-steps of 32)
  const __bf16* Qb = Q + ((size_t)bh * SEQ + q0) * HEAD_DIM;
  BF16x16 aQ[4];
#pragma unroll
  for (int ds = 0; ds < 4; ++ds) {
    const __bf16* p = Qb + (size_t)rln * HEAD_DIM + ds * 32 + hlf * 16;
    aQ[ds].h[0] = *(const v8bf*)(p);
    aQ[ds].h[1] = *(const v8bf*)(p + 8);
  }

  v8f o[8];
  float rmax[8], rsum[8];
#pragma unroll
  for (int dc = 0; dc < 8; ++dc) o[dc] = zero8();
#pragma unroll
  for (int r = 0; r < 8; ++r) { rmax[r] = -3.0e38f; rsum[r] = 0.f; }

  const float scale = 0.08838834764831845f;     // 1/sqrt(128)
  const int ktiles = 2 * qblk + 2;              // causal: keys <= qblk*64+63

  for (int kt = 0; kt < ktiles; ++kt) {
    const __bf16* Kg = Km + ((size_t)bh * SEQ + kt * 32) * HEAD_DIM;
    const __bf16* Vg = Vm + ((size_t)bh * SEQ + kt * 32) * HEAD_DIM;
    // cooperative loads: K straight, V transposed into LDS
#pragma unroll
    for (int c = 0; c < 4; ++c) {
      const int idx = c * 128 + tid;            // 512 x 8-elem chunks
      ((v8bf*)Kt)[idx] = *(const v8bf*)(Kg + (size_t)idx * 8);
      v8bf vv = *(const v8bf*)(Vg + (size_t)idx * 8);
      const int e = idx * 8, key = e >> 7, d0 = e & (HEAD_DIM - 1);
#pragma unroll
      for (int i = 0; i < 8; ++i) Vt[(d0 + i) * 32 + key] = vv[i];
    }
    __syncthreads();

    // ----- scores: S = Q . K^T (contract over d=128) -----
    v8f s0 = zero8(), s1 = zero8();
#pragma unroll
    for (int ds = 0; ds < 4; ++ds) {
      BF16x16 b0, b1;
      const __bf16* p0 = &Kt[rln * HEAD_DIM + ds * 32 + hlf * 16];
      b0.h[0] = *(const v8bf*)(p0);
      b0.h[1] = *(const v8bf*)(p0 + 8);
      const __bf16* p1 = &Kt[(rln + 16) * HEAD_DIM + ds * 32 + hlf * 16];
      b1.h[0] = *(const v8bf*)(p1);
      b1.h[1] = *(const v8bf*)(p1 + 8);
      s0 = wmma_bf16(aQ[ds].v, b0.v, s0);       // keys kt*32 + 0..15
      s1 = wmma_bf16(aQ[ds].v, b1.v, s1);       // keys kt*32 + 16..31
    }

    // ----- causal mask + online softmax (rowwise over 16 lanes) -----
    float fac[8];
#pragma unroll
    for (int r = 0; r < 8; ++r) {
      const int qrow = q0 + hlf * 8 + r;
      const int kcol = kt * 32 + rln;
      float v0 = (kcol <= qrow) ? s0[r] * scale : -3.0e38f;
      float v1 = (kcol + 16 <= qrow) ? s1[r] * scale : -3.0e38f;
      float t = fmaxf(v0, v1);
#pragma unroll
      for (int off = 1; off < 16; off <<= 1) t = fmaxf(t, __shfl_xor(t, off, 16));
      const float nm = fmaxf(rmax[r], t);
      fac[r] = __expf(rmax[r] - nm);
      rmax[r] = nm;
      const float p0v = __expf(v0 - nm);
      const float p1v = __expf(v1 - nm);
      s0[r] = p0v; s1[r] = p1v;
      float rs = p0v + p1v;
#pragma unroll
      for (int off = 1; off < 16; off <<= 1) rs += __shfl_xor(rs, off, 16);
      rsum[r] = rsum[r] * fac[r] + rs;
    }
#pragma unroll
    for (int dc = 0; dc < 8; ++dc)
#pragma unroll
      for (int r = 0; r < 8; ++r) o[dc][r] *= fac[r];

    // ----- reshape P (C/D layout) -> A-fragment layout via per-wave LDS -----
    __bf16* P = Pb[wave];
#pragma unroll
    for (int r = 0; r < 8; ++r) {
      const int row = hlf * 8 + r;
      P[row * 32 + rln]      = (__bf16)s0[r];
      P[row * 32 + 16 + rln] = (__bf16)s1[r];
    }
    BF16x16 aP;
    {
      const __bf16* p = &P[rln * 32 + hlf * 16];   // wave-local, DS in-order
      aP.h[0] = *(const v8bf*)(p);
      aP.h[1] = *(const v8bf*)(p + 8);
    }

    // ----- O += P . V (contract over 32 keys) -----
#pragma unroll
    for (int dc = 0; dc < 8; ++dc) {
      BF16x16 bV;
      const __bf16* p = &Vt[(dc * 16 + rln) * 32 + hlf * 16];
      bV.h[0] = *(const v8bf*)(p);
      bV.h[1] = *(const v8bf*)(p + 8);
      o[dc] = wmma_bf16(aP.v, bV.v, o[dc]);
    }
    __syncthreads();
  }

  // epilogue: normalize and scatter to [B, S, E] bf16
  const int b = bh >> 4, h = bh & (HEADS - 1);
#pragma unroll
  for (int dc = 0; dc < 8; ++dc)
#pragma unroll
    for (int r = 0; r < 8; ++r) {
      const int qrow = q0 + hlf * 8 + r;
      const int d = dc * 16 + rln;
      const float v = o[dc][r] / rsum[r];
      outA[((size_t)(b * SEQ + qrow)) * EMBED + h * HEAD_DIM + d] = (__bf16)v;
    }
}

// ---------------------------------------------------------------------------
extern "C" void kernel_launch(void* const* d_in, const int* in_sizes, int n_in,
                              void* d_out, int out_size, void* d_ws, size_t ws_size,
                              hipStream_t stream) {
  (void)in_sizes; (void)n_in; (void)out_size; (void)ws_size;
  const float* x  = (const float*)d_in[0];
  const float* Wq = (const float*)d_in[1];
  const float* bq = (const float*)d_in[2];
  const float* Wk = (const float*)d_in[3];
  const float* bk = (const float*)d_in[4];
  const float* Wv = (const float*)d_in[5];
  const float* bv = (const float*)d_in[6];
  const float* Wo = (const float*)d_in[7];
  const float* bo = (const float*)d_in[8];
  float* out = (float*)d_out;

  const size_t nX = (size_t)MTOT * EMBED;       // 8,388,608
  const size_t nW = (size_t)EMBED * EMBED;      // 4,194,304

  char* ws = (char*)d_ws;
  __bf16* xb   = (__bf16*)ws; ws += nX * 2;
  __bf16* wqb  = (__bf16*)ws; ws += nW * 2;
  __bf16* wkb  = (__bf16*)ws; ws += nW * 2;
  __bf16* wvb  = (__bf16*)ws; ws += nW * 2;
  __bf16* wob  = (__bf16*)ws; ws += nW * 2;
  __bf16* Qw   = (__bf16*)ws; ws += nX * 2;     // [B,H,S,D]
  __bf16* Kw   = (__bf16*)ws; ws += nX * 2;
  __bf16* Vw   = (__bf16*)ws; ws += nX * 2;
  __bf16* attn = (__bf16*)ws; ws += nX * 2;     // [B,S,E]

  // fp32 -> bf16 (8 elems / iteration)
  const int nX8 = (int)(nX / 8), nW8 = (int)(nW / 8);
  f32_to_bf16_kernel<<<(nX8 + 255) / 256, 256, 0, stream>>>(x, xb, nX8);
  f32_to_bf16_kernel<<<(nW8 + 255) / 256, 256, 0, stream>>>(Wq, wqb, nW8);
  f32_to_bf16_kernel<<<(nW8 + 255) / 256, 256, 0, stream>>>(Wk, wkb, nW8);
  f32_to_bf16_kernel<<<(nW8 + 255) / 256, 256, 0, stream>>>(Wv, wvb, nW8);
  f32_to_bf16_kernel<<<(nW8 + 255) / 256, 256, 0, stream>>>(Wo, wob, nW8);

  // QKV projections: (4096/64)*(2048/64) = 2048 wave-tiles / 4 waves per block
  const int gemmBlocks = (MTOT / 64) * (EMBED / 64) / 4;  // 512
  gemm_bf16_nt<<<gemmBlocks, 128, 0, stream>>>(xb, wqb, bq, Qw, MTOT, EMBED, EMBED, 0);
  gemm_bf16_nt<<<gemmBlocks, 128, 0, stream>>>(xb, wkb, bk, Kw, MTOT, EMBED, EMBED, 0);
  gemm_bf16_nt<<<gemmBlocks, 128, 0, stream>>>(xb, wvb, bv, Vw, MTOT, EMBED, EMBED, 0);

  // causal flash attention
  flash_attn<<<dim3(SEQ / 64, BATCH * HEADS), 128, 0, stream>>>(Qw, Kw, Vw, attn);

  // output projection -> fp32 result
  gemm_bf16_nt<<<gemmBlocks, 128, 0, stream>>>(attn, wob, bo, out, MTOT, EMBED, EMBED, 1);
}
